// SpanClassifier_27436251087029
// MI455X (gfx1250) — compile-verified, matrix operand order checked by reading, not run on hardware
//
#include <hip/hip_runtime.h>

// ---------------------------------------------------------------------------
// SpanClassifier relative-position attention logits for MI455X (gfx1250).
// bf16 WMMA (v_wmma_f32_16x16x32_bf16) + fp32 accumulation.
// Double-buffered async global->LDS staging (ASYNCcnt) feeds ds_load_b128
// fragment reads; fused rel-position gather + mask epilogue writes fp32.
// ---------------------------------------------------------------------------

typedef __bf16 bf16_t;
typedef __attribute__((ext_vector_type(16))) __bf16 v16bf;
typedef __attribute__((ext_vector_type(8)))  __bf16 v8bf;
typedef __attribute__((ext_vector_type(8)))  float  v8f;
typedef __attribute__((ext_vector_type(4)))  int    v4i;

union Frag { v16bf v; v8bf h[2]; };

static constexpr int kB    = 4;
static constexpr int kT    = 2048;
static constexpr int kD    = 1024;
static constexpr int kR    = 64;
static constexpr int kNR   = 2 * kR + 1;   // 129 relative offsets
static constexpr int kQRld = 256;          // padded rel-row count / qr stride
static constexpr int kM    = kB * kT;      // 8192 total rows

static constexpr int kLdsStride = 72;      // 64 elems + 8 pad (bank spread), 144B rows

// ---------------------------------------------------------------------------
// Async global->LDS helpers (guarded; fall back to load+ds_store)
// ---------------------------------------------------------------------------

#if __has_builtin(__builtin_amdgcn_global_load_async_to_lds_b128)
#define HAS_ASYNC_LDS 1
#else
#define HAS_ASYNC_LDS 0
#endif

__device__ __forceinline__ void async_copy16(const bf16_t* gsrc, bf16_t* ldst) {
#if HAS_ASYNC_LDS
  __builtin_amdgcn_global_load_async_to_lds_b128(
      (__attribute__((address_space(1))) v4i*)(gsrc),
      (__attribute__((address_space(3))) v4i*)(ldst), 0, 0);
#else
  *(uint4*)ldst = *(const uint4*)gsrc;
#endif
}

__device__ __forceinline__ void wait_async_le8() {
#if HAS_ASYNC_LDS
#if __has_builtin(__builtin_amdgcn_s_wait_asynccnt)
  __builtin_amdgcn_s_wait_asynccnt(8);
#else
  asm volatile("s_wait_asynccnt 0x8" ::: "memory");
#endif
#endif
}

__device__ __forceinline__ void wait_async_le0() {
#if HAS_ASYNC_LDS
#if __has_builtin(__builtin_amdgcn_s_wait_asynccnt)
  __builtin_amdgcn_s_wait_asynccnt(0);
#else
  asm volatile("s_wait_asynccnt 0x0" ::: "memory");
#endif
#endif
}

// ---------------------------------------------------------------------------
// Prep kernels
// ---------------------------------------------------------------------------

__global__ void k_cvt_bf16(const float* __restrict__ in, bf16_t* __restrict__ out, int n) {
  int i = blockIdx.x * blockDim.x + threadIdx.x;
  if (i < n) out[i] = (bf16_t)in[i];
}

// out[n*D + k] = in[k*D + n]  (so WMMA B-operand reads contiguous K per lane)
__global__ void k_transpose_cvt(const float* __restrict__ in, bf16_t* __restrict__ out) {
  int i = blockIdx.x * blockDim.x + threadIdx.x;   // over D*D
  int k = i >> 10;
  int n = i & (kD - 1);
  out[(size_t)n * kD + k] = (bf16_t)in[i];
}

// relp[256, D] bf16, rows >= 129 zero-padded
__global__ void k_relpad(const float* __restrict__ rel, bf16_t* __restrict__ out) {
  int i = blockIdx.x * blockDim.x + threadIdx.x;   // over 256*D
  int r = i >> 10;
  out[i] = (r < kNR) ? (bf16_t)rel[i] : (bf16_t)0.0f;
}

// ---------------------------------------------------------------------------
// Core 128x128 tile: C = A[M,K] * B[N,K]^T (both row-major, K contiguous).
// Block = 256 threads = 8 waves arranged 4(M) x 2(N); wave tile 32x64.
// Double-buffered 128x64 LDS slabs filled by async b128 copies; fragments
// served from LDS (ds_load_b128), 16 WMMAs per 64-deep k-tile.
// k-tile loop unrolled by exactly the buffer period (2) so `cur` folds to a
// constant per phase without the full-unroll address-ladder blowup.
// ---------------------------------------------------------------------------

__device__ __forceinline__ void gemm_core_128x128(
    const bf16_t* __restrict__ A, int lda, int m0,
    const bf16_t* __restrict__ Bm, int ldb, int n0,
    int K, v8f acc[2][4]) {
  __shared__ bf16_t sA[2][128 * kLdsStride];
  __shared__ bf16_t sB[2][128 * kLdsStride];

  const int tid  = threadIdx.x;
  const int lane = tid & 31;
  const int wave = tid >> 5;
  const int wm = wave >> 1, wn = wave & 1;

  // Loop-invariant divergent byte offsets (4 16B chunks per operand per thread).
  int gA_off[4], gB_off[4], l_off[4];
#pragma unroll
  for (int c = 0; c < 4; ++c) {
    const int q   = tid + c * 256;
    const int row = q >> 3;
    const int col = (q & 7) * 8;
    gA_off[c] = (row * lda + col) * (int)sizeof(bf16_t);
    gB_off[c] = (row * ldb + col) * (int)sizeof(bf16_t);
    l_off[c]  = (row * kLdsStride + col) * (int)sizeof(bf16_t);
  }

  auto issue_tile = [&](int kt, int buf) {
    const char* Ak = (const char*)A  + ((size_t)m0 * lda + kt * 64) * sizeof(bf16_t);
    const char* Bk = (const char*)Bm + ((size_t)n0 * ldb + kt * 64) * sizeof(bf16_t);
    char* sAb = (char*)&sA[buf][0];
    char* sBb = (char*)&sB[buf][0];
#pragma unroll
    for (int c = 0; c < 4; ++c) {
      async_copy16((const bf16_t*)(Ak + gA_off[c]), (bf16_t*)(sAb + l_off[c]));
      async_copy16((const bf16_t*)(Bk + gB_off[c]), (bf16_t*)(sBb + l_off[c]));
    }
  };

  const int NT = K / 64;
  issue_tile(0, 0);

#pragma unroll 2
  for (int it = 0; it < NT; ++it) {
    const int cur = it & 1;
    if (it + 1 < NT) {
      issue_tile(it + 1, cur ^ 1);   // prev compute on this buf ended w/ barrier
      wait_async_le8();              // my 8 newest (next tile) may stay in flight
    } else {
      wait_async_le0();
    }
    __syncthreads();                 // all waves' fills for `cur` complete

#pragma unroll
    for (int kb = 0; kb < 64; kb += 32) {
      const int koff = kb + ((lane >> 4) << 3);   // lanes 16-31: +8 K-half
      Frag a[2];
#pragma unroll
      for (int i = 0; i < 2; ++i) {
        const bf16_t* p = &sA[cur][(wm * 32 + i * 16 + (lane & 15)) * kLdsStride + koff];
        a[i].h[0] = *(const v8bf*)p;
        a[i].h[1] = *(const v8bf*)(p + 16);
      }
      Frag bfr[4];
#pragma unroll
      for (int j = 0; j < 4; ++j) {
        const bf16_t* p = &sB[cur][(wn * 64 + j * 16 + (lane & 15)) * kLdsStride + koff];
        bfr[j].h[0] = *(const v8bf*)p;
        bfr[j].h[1] = *(const v8bf*)(p + 16);
      }
#pragma unroll
      for (int i = 0; i < 2; ++i)
#pragma unroll
        for (int j = 0; j < 4; ++j)
          acc[i][j] = __builtin_amdgcn_wmma_f32_16x16x32_bf16(
              false, a[i].v, false, bfr[j].v, (short)0, acc[i][j], false, false);
    }
    __syncthreads();                 // protect buf before next overwrite
  }
}

// ---------------------------------------------------------------------------
// Projection GEMMs: z=0 Qst, z=1 Kst, z=2 Qed, z=3 Ked
// out_bf16 = (x @ W + b) * scale   (scale = 1/sqrt(D) for Q, 1 for K)
// ---------------------------------------------------------------------------

struct ProjArgs {
  const bf16_t* x;
  const bf16_t* Wt[4];
  const float*  bias[4];
  bf16_t*       out[4];
};

__global__ void __launch_bounds__(256, 2) k_proj_gemm(ProjArgs args) {
  const int z    = blockIdx.z;
  const int lane = threadIdx.x & 31;
  const int wave = threadIdx.x >> 5;
  const int wm = wave >> 1, wn = wave & 1;
  const int m0 = blockIdx.x * 128;
  const int n0 = blockIdx.y * 128;

  v8f acc[2][4];
  v8f zero = {};
#pragma unroll
  for (int i = 0; i < 2; ++i)
#pragma unroll
    for (int j = 0; j < 4; ++j) acc[i][j] = zero;

  gemm_core_128x128(args.x, kD, m0, args.Wt[z], kD, n0, kD, acc);

  const float scale = (z & 1) ? 1.0f : 0.03125f;   // 1/sqrt(1024)
  const float* __restrict__ bias = args.bias[z];
  bf16_t* __restrict__ out = args.out[z];

  const int mhalf = (lane >> 4) * 8;
  const int ncol  = lane & 15;
#pragma unroll
  for (int i = 0; i < 2; ++i) {
#pragma unroll
    for (int j = 0; j < 4; ++j) {
      const int nbase = n0 + wn * 64 + j * 16 + ncol;
      const float bv = bias[nbase];
#pragma unroll
      for (int v = 0; v < 8; ++v) {
        const int m = m0 + wm * 32 + i * 16 + v + mhalf;
        out[(size_t)m * kD + nbase] = (bf16_t)((acc[i][j][v] + bv) * scale);
      }
    }
  }
}

// ---------------------------------------------------------------------------
// qr GEMM: qr[m, r] = Q[m, :] . relp[r, :]   (M=8192, N=256 padded, K=1024)
// ---------------------------------------------------------------------------

__global__ void __launch_bounds__(256, 2) k_qr_gemm(
    const bf16_t* __restrict__ Q0, const bf16_t* __restrict__ Q1,
    const bf16_t* __restrict__ R0, const bf16_t* __restrict__ R1,
    float* __restrict__ qr0, float* __restrict__ qr1) {
  const int h = blockIdx.z;
  const bf16_t* Q  = h ? Q1 : Q0;
  const bf16_t* Rp = h ? R1 : R0;
  float* qr        = h ? qr1 : qr0;

  const int lane = threadIdx.x & 31;
  const int wave = threadIdx.x >> 5;
  const int wm = wave >> 1, wn = wave & 1;
  const int m0 = blockIdx.x * 128;
  const int n0 = blockIdx.y * 128;

  v8f acc[2][4];
  v8f zero = {};
#pragma unroll
  for (int i = 0; i < 2; ++i)
#pragma unroll
    for (int j = 0; j < 4; ++j) acc[i][j] = zero;

  gemm_core_128x128(Q, kD, m0, Rp, kD, n0, kD, acc);

  const int mhalf = (lane >> 4) * 8;
  const int ncol  = lane & 15;
#pragma unroll
  for (int i = 0; i < 2; ++i)
#pragma unroll
    for (int j = 0; j < 4; ++j) {
      const int n = n0 + wn * 64 + j * 16 + ncol;
#pragma unroll
      for (int v = 0; v < 8; ++v) {
        const int m = m0 + wm * 32 + i * 16 + v + mhalf;
        qr[(size_t)m * kQRld + n] = acc[i][j][v];
      }
    }
}

// ---------------------------------------------------------------------------
// Score GEMM with fused relative-position gather + mask epilogue.
// grid.z = head*4 + b ; each block does a 128x128 tile of [T,T].
// ---------------------------------------------------------------------------

__global__ void __launch_bounds__(256, 2) k_scores(
    const bf16_t* __restrict__ Q0, const bf16_t* __restrict__ K0,
    const float* __restrict__ qr0,
    const bf16_t* __restrict__ Q1, const bf16_t* __restrict__ K1,
    const float* __restrict__ qr1,
    const int* __restrict__ mask, float* __restrict__ out) {
  const int z = blockIdx.z;
  const int h = z >> 2;
  const int b = z & 3;

  const bf16_t* Q  = (h ? Q1 : Q0) + (size_t)b * kT * kD;
  const bf16_t* Km = (h ? K1 : K0) + (size_t)b * kT * kD;
  const float*  qr = (h ? qr1 : qr0) + (size_t)b * kT * kQRld;
  const int*    mk = mask + (size_t)b * kT * kT;
  float*        o  = out + (size_t)h * kB * kT * kT + (size_t)b * kT * kT;

  const int lane = threadIdx.x & 31;
  const int wave = threadIdx.x >> 5;
  const int wm = wave >> 1, wn = wave & 1;
  const int m0 = blockIdx.x * 128;
  const int n0 = blockIdx.y * 128;

  v8f acc[2][4];
  v8f zero = {};
#pragma unroll
  for (int i = 0; i < 2; ++i)
#pragma unroll
    for (int j = 0; j < 4; ++j) acc[i][j] = zero;

  gemm_core_128x128(Q, kD, m0, Km, kD, n0, kD, acc);

  const int mhalf = (lane >> 4) * 8;
  const int ncol  = lane & 15;
#pragma unroll
  for (int i = 0; i < 2; ++i) {
#pragma unroll
    for (int j = 0; j < 4; ++j) {
      const int n = n0 + wn * 64 + j * 16 + ncol;
#pragma unroll
      for (int v = 0; v < 8; ++v) {
        const int m = m0 + wm * 32 + i * 16 + v + mhalf;
        int d = n - m;
        d = d < -kR ? -kR : (d > kR ? kR : d);
        const int ridx = d + kR;
        float val = acc[i][j][v] + qr[(size_t)m * kQRld + ridx];
        if (mk[(size_t)m * kT + n] == 0) val = -1e18f;
        o[(size_t)m * kT + n] = val;
      }
    }
  }
}

// ---------------------------------------------------------------------------
// Host launch
// ---------------------------------------------------------------------------

extern "C" void kernel_launch(void* const* d_in, const int* in_sizes, int n_in,
                              void* d_out, int out_size, void* d_ws, size_t ws_size,
                              hipStream_t stream) {
  (void)in_sizes; (void)n_in; (void)out_size; (void)ws_size;

  const float* repre = (const float*)d_in[0];
  const int*   mask  = (const int*)d_in[1];
  const float* Wq[2] = {(const float*)d_in[2], (const float*)d_in[7]};
  const float* bq[2] = {(const float*)d_in[3], (const float*)d_in[8]};
  const float* Wk[2] = {(const float*)d_in[4], (const float*)d_in[9]};
  const float* bk[2] = {(const float*)d_in[5], (const float*)d_in[10]};
  const float* rel[2] = {(const float*)d_in[6], (const float*)d_in[11]};

  // ---- workspace carve-up (≈105 MB) ----
  char* w = (char*)d_ws;
  bf16_t* xb = (bf16_t*)w;               w += (size_t)kM * kD * sizeof(bf16_t);
  bf16_t* Wt[4];
  for (int i = 0; i < 4; ++i) { Wt[i] = (bf16_t*)w; w += (size_t)kD * kD * sizeof(bf16_t); }
  bf16_t* relp[2];
  for (int i = 0; i < 2; ++i) { relp[i] = (bf16_t*)w; w += (size_t)kQRld * kD * sizeof(bf16_t); }
  bf16_t* QK[4];   // Qst, Kst, Qed, Ked  (bf16 [8192,1024])
  for (int i = 0; i < 4; ++i) { QK[i] = (bf16_t*)w; w += (size_t)kM * kD * sizeof(bf16_t); }
  float* qr[2];    // fp32 [8192,256]
  for (int i = 0; i < 2; ++i) { qr[i] = (float*)w; w += (size_t)kM * kQRld * sizeof(float); }

  const int threads = 256;

  // 1) cast x -> bf16
  {
    const int n = kM * kD;
    k_cvt_bf16<<<(n + threads - 1) / threads, threads, 0, stream>>>(repre, xb, n);
  }
  // 2) transpose-cast the 4 weight matrices
  {
    const int n = kD * kD;
    const float* Ws[4] = {Wq[0], Wk[0], Wq[1], Wk[1]};
    for (int i = 0; i < 4; ++i)
      k_transpose_cvt<<<(n + threads - 1) / threads, threads, 0, stream>>>(Ws[i], Wt[i]);
  }
  // 3) pad + cast rel embeddings
  {
    const int n = kQRld * kD;
    for (int i = 0; i < 2; ++i)
      k_relpad<<<(n + threads - 1) / threads, threads, 0, stream>>>(rel[i], relp[i]);
  }
  // 4) fused projection GEMMs (Qst,Kst,Qed,Ked)
  {
    ProjArgs pa;
    pa.x = xb;
    pa.Wt[0] = Wt[0]; pa.Wt[1] = Wt[1]; pa.Wt[2] = Wt[2]; pa.Wt[3] = Wt[3];
    pa.bias[0] = bq[0]; pa.bias[1] = bk[0]; pa.bias[2] = bq[1]; pa.bias[3] = bk[1];
    pa.out[0] = QK[0]; pa.out[1] = QK[1]; pa.out[2] = QK[2]; pa.out[3] = QK[3];
    dim3 grid(kM / 128, kD / 128, 4);
    k_proj_gemm<<<grid, threads, 0, stream>>>(pa);
  }
  // 5) qr = Q @ rel^T (both heads)
  {
    dim3 grid(kM / 128, kQRld / 128, 2);
    k_qr_gemm<<<grid, threads, 0, stream>>>(QK[0], QK[2], relp[0], relp[1], qr[0], qr[1]);
  }
  // 6) scores = Q @ K^T + gathered rel bias, masked, fp32 to d_out
  {
    dim3 grid(kT / 128, kT / 128, kB * 2);
    k_scores<<<grid, threads, 0, stream>>>(QK[0], QK[1], qr[0],
                                           QK[2], QK[3], qr[1],
                                           mask, (float*)d_out);
  }
}
